// DCTAttention_14534169330004
// MI455X (gfx1250) — matrix operand order repaired and checked by reading, not compiled, not run on hardware
//
#include <hip/hip_runtime.h>
#include <math.h>

// ---------------------------------------------------------------------------
// DCT attention for MI455X (gfx1250, wave32, WMMA bf16 16x16x32, f32 accum)
// B=4 H=16 S=4096 D=64, M(freqs)=1024
// GEMM A-tiles streamed via Tensor Data Mover (TDM) with LDS padding,
// double-buffered against the WMMA pipe.
// ---------------------------------------------------------------------------

#define BB 4
#define HH 16
#define SS 4096
#define DD 64
#define MM 1024

typedef __attribute__((ext_vector_type(16))) __bf16 v16bf;
typedef __attribute__((ext_vector_type(8)))  __bf16 v8bf;
typedef __attribute__((ext_vector_type(8)))  float  v8f;
typedef __attribute__((ext_vector_type(4)))  float  v4f;
typedef __attribute__((ext_vector_type(4)))  unsigned int u32x4;
typedef __attribute__((ext_vector_type(8)))  int    i32x8;
typedef __attribute__((ext_vector_type(4)))  int    i32x4;

__device__ __forceinline__ v16bf cat8(v8bf lo, v8bf hi) {
    return __builtin_shufflevector(lo, hi, 0,1,2,3,4,5,6,7,8,9,10,11,12,13,14,15);
}

// ---------------------------------------------------------------------------
// Prep: DCT matrix -> bf16 (and transposed copy)
// ---------------------------------------------------------------------------
__global__ __launch_bounds__(256)
void prep_dct_kernel(const float* __restrict__ D, __bf16* __restrict__ D16,
                     __bf16* __restrict__ Dt16) {
    size_t i = (size_t)blockIdx.x * 256 + threadIdx.x;   // over MM*SS
    int m = (int)(i >> 12);          // row (freq), SS = 4096 = 2^12
    int n = (int)(i & (SS - 1));     // col (time)
    float v = D[i];
    D16[i] = (__bf16)v;
    Dt16[(size_t)n * MM + m] = (__bf16)v;
}

// ---------------------------------------------------------------------------
// Prep: Q*s, K*s*mask, V*mask -> bf16
// ---------------------------------------------------------------------------
__global__ __launch_bounds__(256)
void prep_qkv_kernel(const float* __restrict__ Q, const float* __restrict__ K,
                     const float* __restrict__ V, const float* __restrict__ mask,
                     __bf16* __restrict__ Qh, __bf16* __restrict__ Kh,
                     __bf16* __restrict__ Vh, float s) {
    size_t i = (size_t)blockIdx.x * 256 + threadIdx.x;   // over B*H*S*D
    size_t bh = i >> 18;                 // / (S*D) = / 262144
    int b = (int)(bh >> 4);              // / H
    int n = (int)((i >> 6) & (SS - 1));  // sequence index
    float mv = mask[(size_t)b * SS + n]; // mask[b,0,n]
    Qh[i] = (__bf16)(Q[i] * s);
    Kh[i] = (__bf16)(K[i] * s * mv);
    Vh[i] = (__bf16)(V[i] * mv);
}

// ---------------------------------------------------------------------------
// Generic GEMM: C[M x 64] = A[M x K] @ B[K x 64]
//   - block = 256 threads (8 waves), computes 128 rows x 64 cols
//   - bf16-A path: A tile DMA'd by the Tensor Data Mover into LDS with
//     hardware row padding (64B rows + 16B pad -> stride-40 bf16 layout),
//     double-buffered, synced with s_wait_tensorcnt.
//   - f32-A path (attention probs): VGPR convert staging.
//   - B tile staged transposed in LDS so each lane's WMMA B-fragment
//     (one column, 16 consecutive K) is a contiguous 32B span (two 16B ds reads).
// ---------------------------------------------------------------------------
template<bool A_F32, bool OUT_F32>
__global__ __launch_bounds__(256)
void gemm_n64_kernel(const void* __restrict__ Av, int lda, size_t strideA,
                     const __bf16* __restrict__ Bm, size_t strideB,
                     void* __restrict__ Cv, size_t strideC, int K) {
    __shared__ __align__(16) __bf16 lds_a[2][128][40];   // 128 x 32 K (+8 pad)
    __shared__ __align__(16) __bf16 lds_bT[2][64][40];   // 64 cols x 32 K (+pad)

    const int tid  = threadIdx.x;
    const int lane = tid & 31;
    const int wv   = tid >> 5;       // wave 0..7 -> 16-row strip
    const int r    = lane & 15;
    const int hf   = lane >> 4;      // lane half selects K sub-range
    const int m0   = blockIdx.x * 128;
    const size_t batch = blockIdx.y;

    const __bf16* Ab = A_F32 ? nullptr : ((const __bf16*)Av + strideA * batch);
    const float*  Af = A_F32 ? ((const float*)Av + strideA * batch) : nullptr;
    const __bf16* Bb = Bm + strideB * batch;

    const int bk = tid >> 3, bcs = tid & 7;   // B staging coords

    // ---- stage B tile (transposed): 32 x 64 -> [col][k] ----
    auto stage_b = [&](int buf, int k0) {
        v8bf bv = *(const v8bf*)(Bb + (size_t)(k0 + bk) * 64 + bcs * 8);
        #pragma unroll
        for (int j = 0; j < 8; ++j) lds_bT[buf][bcs * 8 + j][bk] = bv[j];
    };

    // ---- stage A tile: 128 x 32 ----
    auto stage_a = [&](int buf, int k0) {
        if constexpr (A_F32) {
            #pragma unroll
            for (int it = 0; it < 2; ++it) {
                int c = tid + it * 256;            // 0..511 chunks of 8 elems
                int row = c >> 2, seg = c & 3;
                const float* src = Af + (size_t)(m0 + row) * lda + k0 + seg * 8;
                v4f f0 = *(const v4f*)src;
                v4f f1 = *(const v4f*)(src + 4);
                __bf16* dst = &lds_a[buf][row][seg * 8];
                #pragma unroll
                for (int j = 0; j < 4; ++j) {
                    dst[j] = (__bf16)f0[j]; dst[4 + j] = (__bf16)f1[j];
                }
            }
        } else {
            // One wave issues a TDM 2D tile load:
            //   tile = 32 contiguous bf16 (64B) x 128 rows, row stride = lda.
            //   LDS pad: every 16 DWORDs (one 64B row) insert 4 DWORDs (16B)
            //   -> reproduces the stride-40 __bf16 LDS layout in hardware.
            if (wv == 0) {
                unsigned long long ga =
                    (unsigned long long)(uintptr_t)(Ab + (size_t)m0 * lda + k0);
                unsigned int la = (unsigned int)(uintptr_t)(&lds_a[buf][0][0]);
                unsigned int td0 = (unsigned int)lda;      // tensor dim0 (elems)
                unsigned int td1 = 1u << 20;               // tensor dim1 (rows)
                u32x4 g0;
                g0[0] = 1u;                                // count=1, user mode
                g0[1] = la;                                // lds_addr (bytes)
                g0[2] = (unsigned int)ga;                  // global_addr[31:0]
                g0[3] = (unsigned int)((ga >> 32) & 0x01FFFFFFu) | (2u << 30); // type=2
                i32x8 g1;
                g1[0] = (int)((1u << 16)                   // data_size = 2B
                            | (1u << 20)                   // pad_enable
                            | (3u << 22)                   // pad_interval: 16 DW
                            | (3u << 25));                 // pad_amount: 4 DW
                g1[1] = (int)((td0 & 0xFFFFu) << 16);                        // dim0 lo
                g1[2] = (int)((td0 >> 16) | ((td1 & 0xFFFFu) << 16));        // dim0 hi|dim1 lo
                g1[3] = (int)((td1 >> 16) | (32u << 16));                    // dim1 hi|tile_dim0
                g1[4] = (int)128u;                                           // tile_dim1 (tile_dim2=0)
                g1[5] = (int)(unsigned int)lda;                              // dim0_stride lo
                g1[6] = 0;                                                   // dim0_stride hi|dim1_stride lo
                g1[7] = 0;
                i32x4 g2 = (i32x4)0, g3 = (i32x4)0;
                i32x8 g4 = (i32x8)0;   // extra group (6-arg toolchain form)
                __builtin_amdgcn_tensor_load_to_lds(g0, g1, g2, g3, g4, 0);
            }
        }
    };

    v8f acc0 = {}, acc1 = {}, acc2 = {}, acc3 = {};

    // ---- software pipeline: prologue ----
    stage_b(0, 0);
    stage_a(0, 0);

    int nbuf = 0;
    for (int k0 = 0; k0 < K; k0 += 32) {
        const int cur = nbuf;
        nbuf ^= 1;
        const bool more = (k0 + 32) < K;
        if (more) {
            if (k0 + 64 < K)   // prefetch the B tile after next into L2/L0
                __builtin_prefetch(Bb + (size_t)(k0 + 64) * 64, 0, 0);
            stage_b(nbuf, k0 + 32);
            stage_a(nbuf, k0 + 32);
        }
        if constexpr (!A_F32) {
            // TDM completes in order per wave: allow the just-issued tile to
            // stay in flight, require the current one to be done.
            if (wv == 0) {
                if (more) __builtin_amdgcn_s_wait_tensorcnt(1);
                else      __builtin_amdgcn_s_wait_tensorcnt(0);
            }
        }
        __syncthreads();

        // ---- fragments + WMMA on buffer `cur` ----
        v16bf a = cat8(*(const v8bf*)&lds_a[cur][wv * 16 + r][hf * 16],
                       *(const v8bf*)&lds_a[cur][wv * 16 + r][hf * 16 + 8]);
        v16bf b0 = cat8(*(const v8bf*)&lds_bT[cur][ 0 + r][hf * 16],
                        *(const v8bf*)&lds_bT[cur][ 0 + r][hf * 16 + 8]);
        v16bf b1 = cat8(*(const v8bf*)&lds_bT[cur][16 + r][hf * 16],
                        *(const v8bf*)&lds_bT[cur][16 + r][hf * 16 + 8]);
        v16bf b2 = cat8(*(const v8bf*)&lds_bT[cur][32 + r][hf * 16],
                        *(const v8bf*)&lds_bT[cur][32 + r][hf * 16 + 8]);
        v16bf b3 = cat8(*(const v8bf*)&lds_bT[cur][48 + r][hf * 16],
                        *(const v8bf*)&lds_bT[cur][48 + r][hf * 16 + 8]);
        acc0 = __builtin_amdgcn_wmma_f32_16x16x32_bf16(false, a, false, b0, (short)0, acc0, false, false);
        acc1 = __builtin_amdgcn_wmma_f32_16x16x32_bf16(false, a, false, b1, (short)0, acc1, false, false);
        acc2 = __builtin_amdgcn_wmma_f32_16x16x32_bf16(false, a, false, b2, (short)0, acc2, false, false);
        acc3 = __builtin_amdgcn_wmma_f32_16x16x32_bf16(false, a, false, b3, (short)0, acc3, false, false);
        __syncthreads();   // buffer `cur` is re-staged next iteration
    }

    // ---- epilogue: C/D layout: VGPR v -> row hf*8+v, col r ----
    #pragma unroll
    for (int v = 0; v < 8; ++v) {
        size_t row = (size_t)m0 + wv * 16 + hf * 8 + v;
        if constexpr (OUT_F32) {
            float* C = (float*)Cv + strideC * batch + row * 64;
            C[ 0 + r] = acc0[v]; C[16 + r] = acc1[v];
            C[32 + r] = acc2[v]; C[48 + r] = acc3[v];
        } else {
            __bf16* C = (__bf16*)Cv + strideC * batch + row * 64;
            C[ 0 + r] = (__bf16)acc0[v]; C[16 + r] = (__bf16)acc1[v];
            C[32 + r] = (__bf16)acc2[v]; C[48 + r] = (__bf16)acc3[v];
        }
    }
}

// ---------------------------------------------------------------------------
// Scores: Sc[b,h] = Qd[1024x64] @ Kd[1024x64]^T  (K-dim = 64, two WMMA steps)
// Kd rows are exactly the B (K^T) fragments -> contiguous global loads.
// Block: 256 thr = 8 waves; block tile 128x128; wave strip 16x128.
// ---------------------------------------------------------------------------
__global__ __launch_bounds__(256)
void scores_kernel(const __bf16* __restrict__ Qd, const __bf16* __restrict__ Kd,
                   float* __restrict__ Sc) {
    const int lane = threadIdx.x & 31;
    const int wv   = threadIdx.x >> 5;
    const int r    = lane & 15;
    const int hf   = lane >> 4;
    const int m0   = blockIdx.x * 128;
    const int n0   = blockIdx.y * 128;
    const size_t batch = blockIdx.z;

    const __bf16* Qb = Qd + batch * (size_t)(MM * DD);
    const __bf16* Kb = Kd + batch * (size_t)(MM * DD);
    float* Sb = Sc + batch * (size_t)MM * MM;

    const int m = m0 + wv * 16;
    v8f acc[8];
    #pragma unroll
    for (int t = 0; t < 8; ++t) acc[t] = (v8f){};

    #pragma unroll
    for (int kk = 0; kk < 64; kk += 32) {
        const __bf16* arow = Qb + (size_t)(m + r) * DD + kk;
        v16bf a = cat8(*(const v8bf*)(arow + hf * 16),
                       *(const v8bf*)(arow + hf * 16 + 8));
        #pragma unroll
        for (int t = 0; t < 8; ++t) {
            const __bf16* brow = Kb + (size_t)(n0 + t * 16 + r) * DD + kk + hf * 16;
            v16bf b = cat8(*(const v8bf*)brow, *(const v8bf*)(brow + 8));
            acc[t] = __builtin_amdgcn_wmma_f32_16x16x32_bf16(false, a, false, b,
                                                             (short)0, acc[t], false, false);
        }
    }

    #pragma unroll
    for (int t = 0; t < 8; ++t)
        #pragma unroll
        for (int v = 0; v < 8; ++v)
            Sb[(size_t)(m + hf * 8 + v) * MM + n0 + t * 16 + r] = acc[t][v];
}

// ---------------------------------------------------------------------------
// Row softmax over 1024 elems, one wave32 per row (32 elems per lane).
// ---------------------------------------------------------------------------
__global__ __launch_bounds__(256)
void softmax_kernel(float* __restrict__ attn) {
    const int lane = threadIdx.x & 31;
    const size_t row = (size_t)blockIdx.x * 8 + (threadIdx.x >> 5);
    float* p = attn + row * MM;

    float vals[32];
    float mx = -INFINITY;
    #pragma unroll
    for (int i = 0; i < 32; ++i) {
        vals[i] = p[lane + i * 32];
        mx = fmaxf(mx, vals[i]);
    }
    #pragma unroll
    for (int off = 16; off > 0; off >>= 1)
        mx = fmaxf(mx, __shfl_xor(mx, off, 32));

    float sum = 0.0f;
    #pragma unroll
    for (int i = 0; i < 32; ++i) {
        vals[i] = __expf(vals[i] - mx);
        sum += vals[i];
    }
    #pragma unroll
    for (int off = 16; off > 0; off >>= 1)
        sum += __shfl_xor(sum, off, 32);

    const float inv = 1.0f / sum;
    #pragma unroll
    for (int i = 0; i < 32; ++i)
        p[lane + i * 32] = vals[i] * inv;
}

// ---------------------------------------------------------------------------
// Launch
// ---------------------------------------------------------------------------
extern "C" void kernel_launch(void* const* d_in, const int* in_sizes, int n_in,
                              void* d_out, int out_size, void* d_ws, size_t ws_size,
                              hipStream_t stream) {
    (void)in_sizes; (void)n_in; (void)out_size; (void)ws_size;

    const float* Q    = (const float*)d_in[0];
    const float* K    = (const float*)d_in[1];
    const float* V    = (const float*)d_in[2];
    const float* mask = (const float*)d_in[3];
    const float* Dct  = (const float*)d_in[4];   // [MM, SS]

    float* out_x    = (float*)d_out;                         // [B,H,S,D]
    float* out_attn = out_x + (size_t)BB * HH * SS * DD;     // [B,H,MM,MM]

    // workspace carve-up (bf16 buffers), ~150 MB total
    char* ws = (char*)d_ws;
    __bf16* D16  = (__bf16*)ws; ws += (size_t)MM * SS * 2;
    __bf16* Dt16 = (__bf16*)ws; ws += (size_t)SS * MM * 2;
    __bf16* Qh   = (__bf16*)ws; ws += (size_t)BB * HH * SS * DD * 2;
    __bf16* Kh   = (__bf16*)ws; ws += (size_t)BB * HH * SS * DD * 2;
    __bf16* Vh   = (__bf16*)ws; ws += (size_t)BB * HH * SS * DD * 2;
    __bf16* Qd   = (__bf16*)ws; ws += (size_t)BB * HH * MM * DD * 2;
    __bf16* Kd   = (__bf16*)ws; ws += (size_t)BB * HH * MM * DD * 2;
    __bf16* Vd   = (__bf16*)ws; ws += (size_t)BB * HH * MM * DD * 2;
    __bf16* O16  = (__bf16*)ws; ws += (size_t)BB * HH * MM * DD * 2;

    const float s = 0.35355339059327373f;   // HEAD_DIM^-0.25

    // 1) prep
    prep_dct_kernel<<<(MM * SS) / 256, 256, 0, stream>>>(Dct, D16, Dt16);
    prep_qkv_kernel<<<(BB * HH * SS * DD) / 256, 256, 0, stream>>>(Q, K, V, mask,
                                                                   Qh, Kh, Vh, s);

    // 2) DCT projections: [1024x4096] @ [4096x64] per (b,h)
    dim3 gProj(MM / 128, BB * HH);
    gemm_n64_kernel<false, false><<<gProj, 256, 0, stream>>>(
        D16, SS, (size_t)0, Qh, (size_t)SS * DD, Qd, (size_t)MM * DD, SS);
    gemm_n64_kernel<false, false><<<gProj, 256, 0, stream>>>(
        D16, SS, (size_t)0, Kh, (size_t)SS * DD, Kd, (size_t)MM * DD, SS);
    gemm_n64_kernel<false, false><<<gProj, 256, 0, stream>>>(
        D16, SS, (size_t)0, Vh, (size_t)SS * DD, Vd, (size_t)MM * DD, SS);

    // 3) scores -> d_out (raw), then in-place softmax
    dim3 gSc(MM / 128, MM / 128, BB * HH);
    scores_kernel<<<gSc, 256, 0, stream>>>(Qd, Kd, out_attn);
    softmax_kernel<<<(BB * HH * MM) / 8, 256, 0, stream>>>(out_attn);

    // 4) O = attn @ Vd : [1024x1024](f32) @ [1024x64]
    gemm_n64_kernel<true, false><<<gProj, 256, 0, stream>>>(
        out_attn, MM, (size_t)MM * MM, Vd, (size_t)MM * DD, O16, (size_t)MM * DD, MM);

    // 5) x = Dct^T @ O : [4096x1024] @ [1024x64] -> f32 out
    dim3 gInv(SS / 128, BB * HH);
    gemm_n64_kernel<false, true><<<gInv, 256, 0, stream>>>(
        Dt16, MM, (size_t)0, O16, (size_t)MM * DD, out_x, (size_t)SS * DD, MM);
}
